// CausalSelfAttention_17815524344225
// MI455X (gfx1250) — matrix operand order repaired and checked by reading, not compiled, not run on hardware
//
#include <hip/hip_runtime.h>
#include <hip/hip_bf16.h>

// ---------------------------------------------------------------------------
// Causal self-attention for MI455X (gfx1250, wave32, WMMA f16 -> f32 acc).
// B=4, L=2048, D=1024, H=16, HD=64.
//
// Phase 0: one-time converts: x -> f16, in_w/out_w -> f16 TRANSPOSED [N][K].
// Phase 1: qkv = x @ in_w + in_b   (WMMA, async-to-LDS, double buffered,
//          64x64 per-wave tiles -> 16 WMMA per 16 LDS fragment loads).
// Phase 2: flash attention, S^T = K@Q^T so online softmax is in-lane;
//          wave-uniform (readfirstlane) branches keep EXEC all-1s while
//          skipping masked/empty diagonal work.
// Phase 3: out = attn @ out_w + out_b.
// ---------------------------------------------------------------------------

typedef __attribute__((ext_vector_type(16))) _Float16 v16h;
typedef __attribute__((ext_vector_type(8)))  _Float16 v8h;
typedef __attribute__((ext_vector_type(8)))  float    v8f;
typedef __attribute__((ext_vector_type(4)))  float    v4f;

#define BATCH  4
#define SEQ    2048
#define DMODEL 1024
#define NHEAD  16
#define HD     64
#define D3     3072

__device__ __forceinline__ v8f wmma_f16(v16h a, v16h b, v8f c) {
    return __builtin_amdgcn_wmma_f32_16x16x32_f16(
        false, a, false, b, (short)0, c, false, false);
}

// Build a v16h fragment from two 16-byte LDS chunks.
__device__ __forceinline__ v16h frag16_2(const _Float16* p0, const _Float16* p1) {
    v8h a = *(const v8h*)p0;
    v8h b = *(const v8h*)p1;
    return __builtin_shufflevector(a, b, 0,1,2,3,4,5,6,7,8,9,10,11,12,13,14,15);
}

// --- CDNA5 async DMA: global -> LDS, 16 bytes per lane, ASYNCcnt-tracked ---
// GLOBAL_LOAD_ASYNC_TO_LDS_B128 vdst, vaddr, off ; LDS dest VGPR holds the LDS
// byte offset == addr[31:0] of the flat __shared__ pointer.
__device__ __forceinline__ void async_copy16(void* lds_dst, const void* gsrc) {
    unsigned loff = (unsigned)(unsigned long long)lds_dst;
    asm volatile("global_load_async_to_lds_b128 %0, %1, off"
                 :: "v"(loff), "v"(gsrc) : "memory");
}
__device__ __forceinline__ void async_wait0() {
    asm volatile("s_wait_asynccnt 0x0" ::: "memory");
}

// ---------------------------------------------------------------------------
// Phase 0 kernels: converts.
// ---------------------------------------------------------------------------
__global__ __launch_bounds__(256)
void convert_f16(const float* __restrict__ src, _Float16* __restrict__ dst,
                 int n4) {
    int i = blockIdx.x * 256 + threadIdx.x;
    if (i < n4) {
        v4f d = *(const v4f*)&src[(size_t)i * 4];
        _Float16* o = &dst[(size_t)i * 4];
        o[0] = (_Float16)d[0]; o[1] = (_Float16)d[1];
        o[2] = (_Float16)d[2]; o[3] = (_Float16)d[3];
    }
}

// src f32 [R][C] row-major -> dst f16 [C][R] row-major (transposed).
__global__ __launch_bounds__(256)
void convert_transpose(const float* __restrict__ src, _Float16* __restrict__ dst,
                       int R, int C) {
    __shared__ float tile[32][33];
    const int bx = blockIdx.x * 32;
    const int by = blockIdx.y * 32;
    const int tx = threadIdx.x & 31, ty = threadIdx.x >> 5;
#pragma unroll
    for (int p = 0; p < 4; ++p)
        tile[ty + p * 8][tx] = src[(size_t)(by + ty + p * 8) * C + bx + tx];
    __syncthreads();
#pragma unroll
    for (int p = 0; p < 4; ++p)
        dst[(size_t)(bx + ty + p * 8) * R + by + tx] =
            (_Float16)tile[tx][ty + p * 8];
}

// ---------------------------------------------------------------------------
// GEMM with bias: C[M,N] = A[M,K] @ Wt[N,K]^T + bias[N].  A, Wt f16.
// 256 threads (8 waves), block tile 256x128, wave grid 4x2, wave tile 64x64
// (4x4 WMMA accumulators). K-step 32, double-buffered async-to-LDS.
// ---------------------------------------------------------------------------
template <typename OT>
__global__ __launch_bounds__(256)
void gemm_bias(const _Float16* __restrict__ A, const _Float16* __restrict__ Wt,
               const float* __restrict__ bias, OT* __restrict__ C,
               int M, int N, int K) {
    __shared__ alignas(16) _Float16 sA[2][256][40];   // [m][k], padded
    __shared__ alignas(16) _Float16 sB[2][128][40];   // [n][k], padded

    const int t   = threadIdx.x;
    const int l   = t & 31;
    const int wv  = t >> 5;
    const int cL  = l & 15;
    const int hi  = (l < 16) ? 0 : 8;     // A-frag K half offset
    const int kbB = (l < 16) ? 0 : 16;    // unused (B frags use hi too)
    const int m0 = blockIdx.y * 256, n0 = blockIdx.x * 128;
    const int wm = wv >> 1, wn = wv & 1;

    v8f acc[4][4] = {};

    auto stage = [&](int kt, int buf) {
        const int k0 = kt * 32;
#pragma unroll
        for (int p = 0; p < 4; ++p) {          // 256x32 A tile
            int lin = (p * 256 + t) * 8;
            int r = lin >> 5, cc = lin & 31;
            async_copy16(&sA[buf][r][cc], &A[(size_t)(m0 + r) * K + k0 + cc]);
        }
#pragma unroll
        for (int p = 0; p < 2; ++p) {          // 128x32 B tile
            int lin = (p * 256 + t) * 8;
            int r = lin >> 5, cc = lin & 31;
            async_copy16(&sB[buf][r][cc], &Wt[(size_t)(n0 + r) * K + k0 + cc]);
        }
    };

    const int NT = K / 32;
    stage(0, 0);
    async_wait0();
    __syncthreads();

    for (int kt = 0; kt < NT; ++kt) {
        const int cur = kt & 1;
        if (kt + 1 < NT) stage(kt + 1, cur ^ 1);   // DMA overlaps WMMA below

        v16h af[4], bf[4];
#pragma unroll
        for (int i = 0; i < 4; ++i) {
            int r = wm * 64 + i * 16 + cL;
            af[i] = frag16_2(&sA[cur][r][hi], &sA[cur][r][16 + hi]);
        }
#pragma unroll
        for (int j = 0; j < 4; ++j) {
            int n = wn * 64 + j * 16 + cL;
            bf[j] = frag16_2(&sB[cur][n][hi], &sB[cur][n][16 + hi]);
        }
#pragma unroll
        for (int i = 0; i < 4; ++i)
#pragma unroll
            for (int j = 0; j < 4; ++j)
                acc[i][j] = wmma_f16(af[i], bf[j], acc[i][j]);

        async_wait0();       // next tile landed in LDS
        __syncthreads();     // all waves done reading 'cur'
    }

    // C layout: lane = column N (cL), VGPR v = row M (v + hi).
#pragma unroll
    for (int j = 0; j < 4; ++j) {
        int n = n0 + wn * 64 + j * 16 + cL;
        float bv = bias[n];
#pragma unroll
        for (int i = 0; i < 4; ++i)
#pragma unroll
            for (int v = 0; v < 8; ++v) {
                int m = m0 + wm * 64 + i * 16 + v + hi;
                C[(size_t)m * N + n] = (OT)(acc[i][j][v] + bv);
            }
    }
}

// ---------------------------------------------------------------------------
// Flash attention. One workgroup = (b, h, 64-query block), 4 waves.
// Wave w owns queries [qb + 16w, qb + 16w + 16).
// Per 32-key tile: S^T = K @ Q^T (4 WMMA), in-lane online softmax,
// O^T += V^T @ P^T (4 WMMA).  K tiles double-buffered with async DMA.
// ---------------------------------------------------------------------------
__global__ __launch_bounds__(128)
void attn_kernel(const _Float16* __restrict__ qkv, _Float16* __restrict__ attn) {
    __shared__ alignas(16) _Float16 sQ[64][72];         // [q][d]
    __shared__ alignas(16) _Float16 sK[2][32][72];      // [k][d]
    __shared__ alignas(16) _Float16 sVt[2][64][40];     // [d][k] (V transposed)
    __shared__ alignas(16) _Float16 sP[4][16][40];      // per-wave [q][k]

    const int t   = threadIdx.x;
    const int w   = t >> 5, l = t & 31;
    const int cL  = l & 15;
    const int hi  = (l < 16) ? 0 : 8;
    const int kbB = (l < 16) ? 0 : 16;
    const int qb  = blockIdx.x * 64;
    const int h   = blockIdx.y;
    const int b   = blockIdx.z;
    const size_t rowbase = (size_t)(b * SEQ) * D3;
    const int hoff = h * HD;

    // Stage Q (64x64) with async DMA.
#pragma unroll
    for (int p = 0; p < 4; ++p) {
        int lin = (p * 128 + t) * 8;
        int q = lin >> 6, d = lin & 63;
        async_copy16(&sQ[q][d],
                     &qkv[rowbase + (size_t)(qb + q) * D3 + hoff + d]);
    }

    // Stage one 32-key K/V tile: K via async DMA, V transposed manually.
    auto stage = [&](int kt, int buf) {
        const int kbase = kt * 32;
#pragma unroll
        for (int p = 0; p < 2; ++p) {
            int lin = (p * 128 + t) * 8;
            int kk = lin >> 6, d = lin & 63;
            size_t row = rowbase + (size_t)(kbase + kk) * D3 + hoff;
            async_copy16(&sK[buf][kk][d], &qkv[row + DMODEL + d]);
            v8h vv = *(const v8h*)&qkv[row + 2 * DMODEL + d];
#pragma unroll
            for (int i = 0; i < 8; ++i) sVt[buf][d + i][kk] = vv[i];
        }
    };

    const int ntile = (qb >> 5) + 2;   // key tiles of 32 covering keys <= qb+63
    stage(0, 0);
    async_wait0();
    __syncthreads();

    // Q^T B-fragments (resident): lane = query column, head-dim K striped.
    const int ql  = w * 16 + cL;
    const int q_g = qb + ql;
    // Wave-uniform scalar copy of this wave's first query (for s_cbranch
    // uniform control flow: WMMA requires EXEC all-1s, so no lane masking).
    const int wq0 = __builtin_amdgcn_readfirstlane(qb + w * 16);
    v16h bq0 = frag16_2(&sQ[ql][hi],      &sQ[ql][16 + hi]);
    v16h bq1 = frag16_2(&sQ[ql][32 + hi], &sQ[ql][48 + hi]);

    v8f o0 = {}, o1 = {}, o2 = {}, o3 = {};
    float m_run = -1e30f, l_run = 0.0f;
    const float scale = 0.125f;   // 1/sqrt(64)

    for (int kt = 0; kt < ntile; ++kt) {
        const int cur = kt & 1;
        const int kbase = kt * 32;
        if (kt + 1 < ntile) stage(kt + 1, cur ^ 1);   // overlap DMA with math

        if (kbase <= wq0 + 15) {       // tile intersects this wave's causal span
            // S^T = K @ Q^T : C layout lane = query, VGPR = key.
            v8f sta = {}, stb = {};
            {
                const int r0 = cL, r1 = 16 + cL;
                v16h a00 = frag16_2(&sK[cur][r0][hi],      &sK[cur][r0][16 + hi]);
                v16h a01 = frag16_2(&sK[cur][r0][32 + hi], &sK[cur][r0][48 + hi]);
                sta = wmma_f16(a00, bq0, sta);
                sta = wmma_f16(a01, bq1, sta);
                v16h a10 = frag16_2(&sK[cur][r1][hi],      &sK[cur][r1][16 + hi]);
                v16h a11 = frag16_2(&sK[cur][r1][32 + hi], &sK[cur][r1][48 + hi]);
                stb = wmma_f16(a10, bq0, stb);
                stb = wmma_f16(a11, bq1, stb);
            }

            float pa[8], pb[8];
            float tmax = -1e30f;
            if (kbase + 31 > wq0) {
                // Diagonal tile: apply causal mask (cndmask path).
#pragma unroll
                for (int v = 0; v < 8; ++v) {
                    int ka = kbase + v + hi;
                    int kb = kbase + 16 + v + hi;
                    pa[v] = (ka <= q_g) ? sta[v] * scale : -1e30f;
                    pb[v] = (kb <= q_g) ? stb[v] * scale : -1e30f;
                    tmax = fmaxf(tmax, fmaxf(pa[v], pb[v]));
                }
            } else {
                // Interior tile: no masking needed.
#pragma unroll
                for (int v = 0; v < 8; ++v) {
                    pa[v] = sta[v] * scale;
                    pb[v] = stb[v] * scale;
                    tmax = fmaxf(tmax, fmaxf(pa[v], pb[v]));
                }
            }
            tmax = fmaxf(tmax, __shfl_xor(tmax, 16));
            float m_new = fmaxf(m_run, tmax);
            float alpha = __expf(m_run - m_new);
            float psum = 0.0f;
#pragma unroll
            for (int v = 0; v < 8; ++v) {
                pa[v] = __expf(pa[v] - m_new);
                pb[v] = __expf(pb[v] - m_new);
                psum += pa[v] + pb[v];
            }
            psum += __shfl_xor(psum, 16);
            l_run = l_run * alpha + psum;
            m_run = m_new;
#pragma unroll
            for (int v = 0; v < 8; ++v) {
                o0[v] *= alpha; o1[v] *= alpha; o2[v] *= alpha; o3[v] *= alpha;
            }

            // Route P^T (C layout) -> B layout through per-wave LDS [q][k].
#pragma unroll
            for (int v = 0; v < 8; ++v) {
                sP[w][cL][v + hi]      = (_Float16)pa[v];
                sP[w][cL][16 + v + hi] = (_Float16)pb[v];
            }
            // DS ops from one wave are in-order; loads below see stores above.
            v16h bp = frag16_2(&sP[w][cL][hi], &sP[w][cL][16 + hi]);

            // O^T += V^T @ P^T  (4 head-dim tiles of 16).
            v16h av;
            av = frag16_2(&sVt[cur][cL][hi],      &sVt[cur][cL][16 + hi]);
            o0 = wmma_f16(av, bp, o0);
            av = frag16_2(&sVt[cur][16 + cL][hi], &sVt[cur][16 + cL][16 + hi]);
            o1 = wmma_f16(av, bp, o1);
            av = frag16_2(&sVt[cur][32 + cL][hi], &sVt[cur][32 + cL][16 + hi]);
            o2 = wmma_f16(av, bp, o2);
            av = frag16_2(&sVt[cur][48 + cL][hi], &sVt[cur][48 + cL][16 + hi]);
            o3 = wmma_f16(av, bp, o3);
        }

        async_wait0();       // next K tile landed (wave still co-stages)
        __syncthreads();     // all waves done with 'cur'
    }

    // Normalize and store [B, L, D] with D = h*64 + d; per-lane dims are
    // contiguous 8-half chunks, 16B aligned -> 4 global_store_b128.
    const float inv = 1.0f / l_run;
    const size_t orow = ((size_t)(b * SEQ) + q_g) * DMODEL + hoff;
    v8h r0, r1, r2, r3;
#pragma unroll
    for (int v = 0; v < 8; ++v) {
        r0[v] = (_Float16)(o0[v] * inv);
        r1[v] = (_Float16)(o1[v] * inv);
        r2[v] = (_Float16)(o2[v] * inv);
        r3[v] = (_Float16)(o3[v] * inv);
    }
    *(v8h*)&attn[orow + 0  + hi] = r0;
    *(v8h*)&attn[orow + 16 + hi] = r1;
    *(v8h*)&attn[orow + 32 + hi] = r2;
    *(v8h*)&attn[orow + 48 + hi] = r3;
}

// ---------------------------------------------------------------------------
extern "C" void kernel_launch(void* const* d_in, const int* in_sizes, int n_in,
                              void* d_out, int out_size, void* d_ws, size_t ws_size,
                              hipStream_t stream) {
    const float* x     = (const float*)d_in[0];   // [B, L, D]
    const float* in_w  = (const float*)d_in[1];   // [D, 3D]
    const float* in_b  = (const float*)d_in[2];   // [3D]
    const float* out_w = (const float*)d_in[3];   // [D, D]
    const float* out_b = (const float*)d_in[4];   // [D]
    float* out = (float*)d_out;                   // [B, L, D]

    const int M = BATCH * SEQ;                    // 8192
    _Float16* qkv_h  = (_Float16*)d_ws;                         // M x 3D
    _Float16* attn_h = qkv_h  + (size_t)M * D3;                 // M x D
    _Float16* x_h    = attn_h + (size_t)M * DMODEL;             // M x D
    _Float16* wt1    = x_h    + (size_t)M * DMODEL;             // 3D x D (in_w^T)
    _Float16* wt2    = wt1    + (size_t)D3 * DMODEL;            // D x D  (out_w^T)

    // Phase 0: converts (one-time; ~75 MB => ~3.2 us at 23.3 TB/s).
    convert_f16<<<(M * DMODEL / 4 + 255) / 256, 256, 0, stream>>>(
        x, x_h, M * DMODEL / 4);
    convert_transpose<<<dim3(D3 / 32, DMODEL / 32), 256, 0, stream>>>(
        in_w, wt1, DMODEL, D3);
    convert_transpose<<<dim3(DMODEL / 32, DMODEL / 32), 256, 0, stream>>>(
        out_w, wt2, DMODEL, DMODEL);

    // Phase 1: QKV projection (block tile 256x128).
    gemm_bias<_Float16><<<dim3(D3 / 128, M / 256), 256, 0, stream>>>(
        x_h, wt1, in_b, qkv_h, M, D3, DMODEL);

    // Phase 2: flash attention.
    attn_kernel<<<dim3(SEQ / 64, NHEAD, BATCH), 128, 0, stream>>>(qkv_h, attn_h);

    // Phase 3: output projection (fp32 out + bias).
    gemm_bias<float><<<dim3(DMODEL / 128, M / 256), 256, 0, stream>>>(
        attn_h, wt2, out_b, out, M, DMODEL, DMODEL);
}
